// LstmChordPredictor_36850819400220
// MI455X (gfx1250) — matrix operand order repaired and checked by reading, not compiled, not run on hardware
//
#include <hip/hip_runtime.h>

#define Hd   1024
#define Gd   4096      // 4*H gate outputs
#define Td   8192
#define NBLK 32
#define NTHR 256       // 8 waves/block, 256 waves total = 256 output tiles

typedef __attribute__((ext_vector_type(16))) __bf16 v16bf;
typedef __attribute__((ext_vector_type(8)))  __bf16 v8bf;
typedef __attribute__((ext_vector_type(8)))  float  v8f;

__device__ __forceinline__ __bf16 f2bf(float f) {
  unsigned u = __float_as_uint(f);
  u += 0x7fffu + ((u >> 16) & 1u);            // round-to-nearest-even
  unsigned short hs = (unsigned short)(u >> 16);
  __bf16 b;
  __builtin_memcpy(&b, &hs, 2);
  return b;
}

__device__ __forceinline__ v16bf cat16(v8bf lo, v8bf hi) {
  return __builtin_shufflevector(lo, hi, 0,1,2,3,4,5,6,7,8,9,10,11,12,13,14,15);
}

// fast activations: v_exp_f32 + v_rcp_f32, no division fixup chains
__device__ __forceinline__ float fast_sigmoid(float x) {
  float e = __expf(-x);
  return __builtin_amdgcn_rcpf(1.f + e);
}
__device__ __forceinline__ float fast_tanh(float x) {
  float ax = fabsf(x);
  float e  = __expf(-2.f * ax);
  float r  = 1.f - 2.f * e * __builtin_amdgcn_rcpf(1.f + e);
  return copysignf(r, x);
}

// -------- init: zero h-state and barrier counters each launch --------
__global__ void init_state(__bf16* h1g, __bf16* h2g, unsigned* bar) {
  int i = blockIdx.x * blockDim.x + threadIdx.x;
  if (i < Hd) {
    unsigned short z = 0;
    __builtin_memcpy(&h1g[i], &z, 2);
    __builtin_memcpy(&h2g[i], &z, 2);
  }
  if (i < 2) bar[i] = 0;
}

// -------- weight conversion: fp32 -> bf16, gate-interleaved row permutation --------
// Permuted row r' = 16*t + s maps to original row (s&3)*H + 4*t + (s>>2):
// a 16-wide output tile t holds i,f,g,o for hidden units 4t..4t+3.
__global__ void __launch_bounds__(256) convert_weights(
    const float* __restrict__ w_hh1, const float* __restrict__ w_ih2,
    const float* __restrict__ w_hh2, const float* __restrict__ w_ih1,
    const float* __restrict__ b_ih1, const float* __restrict__ b_hh1,
    const float* __restrict__ b_ih2, const float* __restrict__ b_hh2,
    __bf16* __restrict__ Wb1, __bf16* __restrict__ Wb2i, __bf16* __restrict__ Wb2h,
    float* __restrict__ w1p, float* __restrict__ b1p, float* __restrict__ b2p)
{
  int rp = blockIdx.x;                       // 0..4095 permuted row
  int tt = rp >> 4, s = rp & 15;
  int orig = (s & 3) * Hd + 4 * tt + (s >> 2);
  const float* s1 = w_hh1 + (size_t)orig * Hd;
  const float* s2 = w_ih2 + (size_t)orig * Hd;
  const float* s3 = w_hh2 + (size_t)orig * Hd;
  __bf16* d1 = Wb1  + (size_t)rp * Hd;
  __bf16* d2 = Wb2i + (size_t)rp * Hd;
  __bf16* d3 = Wb2h + (size_t)rp * Hd;
  for (int c = threadIdx.x; c < Hd; c += blockDim.x) {
    d1[c] = f2bf(s1[c]);
    d2[c] = f2bf(s2[c]);
    d3[c] = f2bf(s3[c]);
  }
  if (threadIdx.x == 0) {
    w1p[rp] = w_ih1[orig];                   // input weight column (x is scalar)
    b1p[rp] = b_ih1[orig] + b_hh1[orig];
    b2p[rp] = b_ih2[orig] + b_hh2[orig];
  }
}

// -------- grid-wide barrier (monotonic generation, reset each launch) --------
__device__ __forceinline__ void grid_sync(unsigned* cnt, unsigned* rel, unsigned gen) {
  __threadfence();
  __syncthreads();
  if (threadIdx.x == 0) {
    unsigned prev = __hip_atomic_fetch_add(cnt, 1u, __ATOMIC_ACQ_REL, __HIP_MEMORY_SCOPE_AGENT);
    if (prev == (unsigned)NBLK * gen - 1u) {
      __hip_atomic_store(rel, gen, __ATOMIC_RELEASE, __HIP_MEMORY_SCOPE_AGENT);
    } else {
      while (__hip_atomic_load(rel, __ATOMIC_ACQUIRE, __HIP_MEMORY_SCOPE_AGENT) < gen)
        __builtin_amdgcn_s_sleep(2);
    }
  }
  __syncthreads();
  __threadfence();
}

// -------- persistent two-layer LSTM over all T steps --------
__global__ void __launch_bounds__(NTHR)
lstm_persistent(const float* __restrict__ seq,
                const __bf16* __restrict__ Wb1,
                const __bf16* __restrict__ Wb2i,
                const __bf16* __restrict__ Wb2h,
                const float* __restrict__ w1p,
                const float* __restrict__ b1p,
                const float* __restrict__ b2p,
                const float* __restrict__ wlin,
                const float* __restrict__ blin,
                __bf16* __restrict__ h1g,
                __bf16* __restrict__ h2g,
                float* __restrict__ pscr,
                unsigned* __restrict__ bar,
                float* __restrict__ out)
{
  __shared__ __bf16 sh1[Hd];
  __shared__ __bf16 sh2[Hd];
  __shared__ float  red[NTHR];

  const int tid  = threadIdx.x;
  const int lane = tid & 31;
  const int wv   = (blockIdx.x << 3) | (tid >> 5);   // 0..255, one 16-col tile per wave
  const int nloc = lane & 15;
  const int row  = (wv << 4) + nloc;                 // permuted gate column owned by lane
  const int kh   = lane >> 4;                        // lane half selects K sub-block

  // B-matrix (32x16 K-major) base pointers: lane l loads 16 contiguous bf16 per K-tile
  const __bf16* WB1  = Wb1  + (size_t)row * Hd + kh * 16;
  const __bf16* WB2i = Wb2i + (size_t)row * Hd + kh * 16;
  const __bf16* WB2h = Wb2h + (size_t)row * Hd + kh * 16;

  const float wv1 = w1p[row];
  const float bb1 = b1p[row];
  const float bb2 = b2p[row];
  const float bL  = blin[0];
  float wlu = 0.f;
  if (lane < 16 && (lane & 3) == 0) wlu = wlin[(wv << 2) + (lane >> 2)];

  float c1 = 0.f, c2 = 0.f;                          // cell state lives in registers
  unsigned gen = 0;
  const int shoff = kh * 16;                         // A-matrix lane-half K offset

  for (int t = 0; t < Td; ++t) {
    // ---- stage h1 (t-1) into LDS ----
    ((uint2*)sh1)[tid] = ((const uint2*)h1g)[tid];
    __syncthreads();

    const float xt = seq[t];

    // ---- layer-1 GEMV: two independent accumulator chains over K=1024 ----
    v8f accA = {}, accB = {};
    #pragma unroll 4
    for (int k0 = 0; k0 < Hd; k0 += 64) {
      v16bf a0 = cat16(*(const v8bf*)(sh1 + shoff + k0),
                       *(const v8bf*)(sh1 + shoff + k0 + 8));
      v16bf b0 = *(const v16bf*)(WB1 + k0);
      accA = __builtin_amdgcn_wmma_f32_16x16x32_bf16(false, a0, false, b0,
                                                     (short)0, accA, false, false);
      v16bf a1 = cat16(*(const v8bf*)(sh1 + shoff + k0 + 32),
                       *(const v8bf*)(sh1 + shoff + k0 + 40));
      v16bf b1 = *(const v16bf*)(WB1 + k0 + 32);
      accB = __builtin_amdgcn_wmma_f32_16x16x32_bf16(false, a1, false, b1,
                                                     (short)0, accB, false, false);
    }

    // ---- layer-1 elementwise: lanes 0..15 hold (i,f,g,o) quads for units 4wv..4wv+3 ----
    {
      float gv = (accA[0] + accB[0]) + xt * wv1 + bb1;
      float av = ((lane & 3) == 2) ? fast_tanh(gv) : fast_sigmoid(gv);
      int  b4 = lane & ~3;
      float gi = __shfl(av, b4 + 0, 32);
      float gf = __shfl(av, b4 + 1, 32);
      float gg = __shfl(av, b4 + 2, 32);
      float go = __shfl(av, b4 + 3, 32);
      c1 = gf * c1 + gi * gg;
      float h1n = go * fast_tanh(c1);
      if (lane < 16 && (lane & 3) == 0)
        h1g[(wv << 2) + (lane >> 2)] = f2bf(h1n);
    }

    grid_sync(bar, bar + 1, ++gen);                  // h1(t) visible device-wide

    // ---- stage h1(t), h2(t-1) into LDS ----
    ((uint2*)sh1)[tid] = ((const uint2*)h1g)[tid];
    ((uint2*)sh2)[tid] = ((const uint2*)h2g)[tid];
    __syncthreads();

    // ---- layer-2 GEMV: gates2 = h1 @ Wih2^T + h2 @ Whh2^T, two independent chains ----
    v8f accI = {}, accH = {};
    #pragma unroll 2
    for (int k0 = 0; k0 < Hd; k0 += 32) {
      v16bf a1 = cat16(*(const v8bf*)(sh1 + shoff + k0),
                       *(const v8bf*)(sh1 + shoff + k0 + 8));
      v16bf bv1 = *(const v16bf*)(WB2i + k0);
      accI = __builtin_amdgcn_wmma_f32_16x16x32_bf16(false, a1, false, bv1,
                                                     (short)0, accI, false, false);
      v16bf a2 = cat16(*(const v8bf*)(sh2 + shoff + k0),
                       *(const v8bf*)(sh2 + shoff + k0 + 8));
      v16bf bv2 = *(const v16bf*)(WB2h + k0);
      accH = __builtin_amdgcn_wmma_f32_16x16x32_bf16(false, a2, false, bv2,
                                                     (short)0, accH, false, false);
    }

    // ---- layer-2 elementwise + partial output projection ----
    float contrib = 0.f;
    {
      float gv = (accI[0] + accH[0]) + bb2;
      float av = ((lane & 3) == 2) ? fast_tanh(gv) : fast_sigmoid(gv);
      int  b4 = lane & ~3;
      float gi = __shfl(av, b4 + 0, 32);
      float gf = __shfl(av, b4 + 1, 32);
      float gg = __shfl(av, b4 + 2, 32);
      float go = __shfl(av, b4 + 3, 32);
      c2 = gf * c2 + gi * gg;
      float h2n = go * fast_tanh(c2);
      if (lane < 16 && (lane & 3) == 0) {
        h2g[(wv << 2) + (lane >> 2)] = f2bf(h2n);
        contrib = h2n * wlu;
      }
    }
    #pragma unroll
    for (int off = 16; off > 0; off >>= 1)
      contrib += __shfl_down(contrib, off, 32);
    if (lane == 0) pscr[wv] = contrib;

    grid_sync(bar, bar + 1, ++gen);                  // h2(t) + partials visible

    // ---- deterministic reduction of 256 wave partials -> out[t] (block 0 only) ----
    if (blockIdx.x == 0) {
      red[tid] = pscr[tid];
      __syncthreads();
      for (int off = NTHR / 2; off > 0; off >>= 1) {
        if (tid < off) red[tid] += red[tid + off];
        __syncthreads();
      }
      if (tid == 0) out[t] = red[0] + bL;
      __syncthreads();
    }
  }
}

extern "C" void kernel_launch(void* const* d_in, const int* in_sizes, int n_in,
                              void* d_out, int out_size, void* d_ws, size_t ws_size,
                              hipStream_t stream) {
  const float* seq   = (const float*)d_in[0];
  const float* w_ih1 = (const float*)d_in[1];
  const float* w_hh1 = (const float*)d_in[2];
  const float* b_ih1 = (const float*)d_in[3];
  const float* b_hh1 = (const float*)d_in[4];
  const float* w_ih2 = (const float*)d_in[5];
  const float* w_hh2 = (const float*)d_in[6];
  const float* b_ih2 = (const float*)d_in[7];
  const float* b_hh2 = (const float*)d_in[8];
  const float* w_lin = (const float*)d_in[9];
  const float* b_lin = (const float*)d_in[10];
  float* out = (float*)d_out;

  // workspace layout (~25.2 MB): three bf16 weight images + vectors + state + barrier
  char* ws = (char*)d_ws;
  const size_t SZW = (size_t)Gd * Hd * sizeof(unsigned short);   // 8 MB each
  __bf16* Wb1  = (__bf16*)(ws);
  __bf16* Wb2i = (__bf16*)(ws + SZW);
  __bf16* Wb2h = (__bf16*)(ws + 2 * SZW);
  float*  w1p  = (float*)(ws + 3 * SZW);
  float*  b1p  = w1p + Gd;
  float*  b2p  = b1p + Gd;
  __bf16* h1g  = (__bf16*)(b2p + Gd);
  __bf16* h2g  = h1g + Hd;
  float*  pscr = (float*)(h2g + Hd);
  unsigned* bar = (unsigned*)(pscr + 256);

  init_state<<<4, 256, 0, stream>>>(h1g, h2g, bar);
  convert_weights<<<Gd, 256, 0, stream>>>(w_hh1, w_ih2, w_hh2, w_ih1,
                                          b_ih1, b_hh1, b_ih2, b_hh2,
                                          Wb1, Wb2i, Wb2h, w1p, b1p, b2p);
  lstm_persistent<<<NBLK, NTHR, 0, stream>>>(seq, Wb1, Wb2i, Wb2h,
                                             w1p, b1p, b2p, w_lin, b_lin,
                                             h1g, h2g, pscr, bar, out);
}
